// FlexInteractionBlock_1778116461345
// MI455X (gfx1250) — compile-verified, matrix operand order checked by reading, not run on hardware
//
#include <hip/hip_runtime.h>
#include <hip/hip_bf16.h>
#include <math.h>

// ---------------------------------------------------------------------------
// Types for CDNA5 WMMA (wave32, 16x16x32 bf16 -> f32)
// ---------------------------------------------------------------------------
typedef __attribute__((ext_vector_type(16))) __bf16 v16bf;
typedef __attribute__((ext_vector_type(8)))  float  v8f;
typedef __attribute__((ext_vector_type(4)))  int    v4i;
typedef __attribute__((ext_vector_type(4)))  float  v4f;

#define MUL   64
#define NRAD  32
#define MLPW  128
#define NMSG  192
#define K1A   161   // actual mlp_in dim
#define K1P   192   // padded to multiple of 32

constexpr int K1_STEPS = K1P / 32;   // 6
constexpr int K2_STEPS = MLPW / 32;  // 4
constexpr int NT_H     = MLPW / 16;  // 8
constexpr int NT_M     = NMSG / 16;  // 12
constexpr int FRAGS_L1 = K1_STEPS * NT_H; // 48
constexpr int FRAGS_L2 = K2_STEPS * NT_H; // 32
constexpr int FRAGS_L3 = K2_STEPS * NT_M; // 48

__device__ __forceinline__ float silu(float x) {
    return x / (1.0f + __expf(-x));
}

union FragU { v16bf v; v4i q[2]; };

// K-local index inside a 16x32 bf16 A (or mirrored B) fragment:
// lanes 0-15 : elements 0..7 -> K 0..7,  elements 8..15 -> K 16..23
// lanes 16-31: elements 0..7 -> K 8..15, elements 8..15 -> K 24..31
__device__ __forceinline__ int klocal_of(int i, int kh) {
    return (i < 8) ? (kh * 8 + i) : (16 + kh * 8 + (i - 8));
}

// Load a pre-swizzled B fragment from global memory (2 x 16B per lane).
__device__ __forceinline__ v16bf load_frag_g(const __bf16* __restrict__ base,
                                             int frag, int lane) {
    const v4i* p = (const v4i*)(base + (size_t)frag * 512 + lane * 16);
    FragU u;
    u.q[0] = p[0];
    u.q[1] = p[1];
    return u.v;
}

// Build an A fragment from a row-major bf16 tile in LDS (rowStride in elems,
// must keep 16B alignment: rowStride*2 % 16 == 0).
__device__ __forceinline__ v16bf load_a_lds(const __bf16* lds, int rowStride,
                                            int kstep, int lane) {
    int lm = lane & 15, kh = lane >> 4;
    const __bf16* p0 = lds + lm * rowStride + kstep * 32 + kh * 8;
    FragU u;
    u.q[0] = *(const v4i*)(p0);
    u.q[1] = *(const v4i*)(p0 + 16);
    return u.v;
}

__device__ __forceinline__ v8f wmma_bf16(v16bf a, v16bf b, v8f c) {
    return __builtin_amdgcn_wmma_f32_16x16x32_bf16(
        /*neg_a=*/false, a, /*neg_b=*/false, b,
        /*c_mod=*/(short)0, c, /*reuse_a=*/false, /*reuse_b=*/false);
}

// ---------------------------------------------------------------------------
// Kernel: convert + swizzle W1/W2/W3 into bf16 fragment-major layout.
// One thread per fragment element. 128 frags * 512 elems = 65536 threads.
// ---------------------------------------------------------------------------
__global__ void prep_weights(const float* __restrict__ W1,
                             const float* __restrict__ W2,
                             const float* __restrict__ W3,
                             __bf16* __restrict__ W1s,
                             __bf16* __restrict__ W2s,
                             __bf16* __restrict__ W3s) {
    int idx = blockIdx.x * blockDim.x + threadIdx.x;
    if (idx >= (FRAGS_L1 + FRAGS_L2 + FRAGS_L3) * 512) return;
    int frag = idx >> 9;
    int pos  = idx & 511;
    int lane = pos >> 4;
    int i    = pos & 15;
    int kh = lane >> 4, lm = lane & 15;
    int kl = klocal_of(i, kh);

    const float* W; __bf16* Wd; int K, N, nt, f;
    if (frag < FRAGS_L1)                { W = W1; Wd = W1s; K = K1A;  N = MLPW; nt = NT_H; f = frag; }
    else if (frag < FRAGS_L1+FRAGS_L2)  { W = W2; Wd = W2s; K = MLPW; N = MLPW; nt = NT_H; f = frag - FRAGS_L1; }
    else                                { W = W3; Wd = W3s; K = MLPW; N = NMSG; nt = NT_M; f = frag - FRAGS_L1 - FRAGS_L2; }

    int k = f / nt, n = f % nt;
    int kg  = k * 32 + kl;
    int col = n * 16 + lm;
    float val = (kg < K) ? W[(size_t)kg * N + col] : 0.0f;
    Wd[(size_t)f * 512 + pos] = (__bf16)val;
}

__global__ void zero_kernel(float* __restrict__ p, size_t n) {
    size_t i = (size_t)blockIdx.x * blockDim.x + threadIdx.x;
    if (i < n) p[i] = 0.0f;
}

// ---------------------------------------------------------------------------
// Node up-projection: s_up = ns @ W_up_s + b ; v_up[n,d,i] = sum_c nv[n,c,i]*Wv[c,d]
// Block = 4 nodes x 64 channels.
// ---------------------------------------------------------------------------
__global__ void __launch_bounds__(256)
up_kernel(const float* __restrict__ ns, const float* __restrict__ nv,
          const float* __restrict__ Ws, const float* __restrict__ bs,
          const float* __restrict__ Wv,
          float* __restrict__ s_up, float* __restrict__ v_up, int nNodes) {
    __shared__ float sN[4][MUL];
    __shared__ float sV[4][MUL][3];
    int ln  = threadIdx.x & 63;
    int sub = threadIdx.x >> 6;
    int node = blockIdx.x * 4 + sub;
    bool ok = node < nNodes;
    if (ok) {
        sN[sub][ln] = ns[(size_t)node * MUL + ln];
        sV[sub][ln][0] = nv[((size_t)node * MUL + ln) * 3 + 0];
        sV[sub][ln][1] = nv[((size_t)node * MUL + ln) * 3 + 1];
        sV[sub][ln][2] = nv[((size_t)node * MUL + ln) * 3 + 2];
    }
    __syncthreads();
    if (!ok) return;
    float accS = bs[ln];
    float a0 = 0.f, a1 = 0.f, a2 = 0.f;
    for (int c = 0; c < MUL; ++c) {
        float ws = Ws[c * MUL + ln];
        float wv = Wv[c * MUL + ln];
        accS += sN[sub][c] * ws;
        a0 += sV[sub][c][0] * wv;
        a1 += sV[sub][c][1] * wv;
        a2 += sV[sub][c][2] * wv;
    }
    s_up[(size_t)node * MUL + ln] = accS;
    v_up[((size_t)node * MUL + ln) * 3 + 0] = a0;
    v_up[((size_t)node * MUL + ln) * 3 + 1] = a1;
    v_up[((size_t)node * MUL + ln) * 3 + 2] = a2;
}

// ---------------------------------------------------------------------------
// Edge kernel: one wave32 per 16-edge tile. Three WMMA MLP layers + messages.
// ---------------------------------------------------------------------------
struct alignas(16) WaveSlice {
    __bf16 buf[16 * K1P];   // 16 x 192 bf16 tile (reused between layers)
    int    snd[16];
    int    rcv[16];
    float  ev[16][3];
};

__global__ void __launch_bounds__(256)
edge_kernel(const int* __restrict__ senders, const int* __restrict__ receivers,
            const float* __restrict__ evec, const float* __restrict__ radial,
            const float* __restrict__ lengths,
            const float* __restrict__ s_up, const float* __restrict__ v_up,
            const __bf16* __restrict__ W1s, const __bf16* __restrict__ W2s,
            const __bf16* __restrict__ W3s,
            const float* __restrict__ B1, const float* __restrict__ B2,
            const float* __restrict__ B3,
            float* __restrict__ agg0, float* __restrict__ agg1, int nEdges) {
    __shared__ WaveSlice slices[8];

    const int wave = threadIdx.x >> 5;
    const int lane = threadIdx.x & 31;
    const int tile = blockIdx.x * 8 + wave;      // wave-uniform
    if (tile * 16 >= nEdges) return;             // uniform exit, EXEC stays full
    WaveSlice& S = slices[wave];
    const int lm = lane & 15, lh = lane >> 4;

    // ---- Phase 1: edge metadata -------------------------------------------
    if (lane < 16) {
        int e = tile * 16 + lane;
        int es = (e < nEdges) ? e : 0;
        S.snd[lane] = senders[es];
        S.rcv[lane] = receivers[es];
        S.ev[lane][0] = evec[(size_t)es * 3 + 0];
        S.ev[lane][1] = evec[(size_t)es * 3 + 1];
        S.ev[lane][2] = evec[(size_t)es * 3 + 2];
    }
    __builtin_amdgcn_wave_barrier();   // keep DS order (in-order LDS per wave)

    // ---- Phase 2: build mlp_in tile (16 x 192 bf16) -----------------------
    {
        int e  = tile * 16 + lm;
        int es = (e < nEdges) ? e : 0;
        __bf16* row = S.buf + lm * K1P;
        if (lh == 0) {
            // cols 0..63 : sender scalars ; cols 128..159 : radial feats
            const v4f* srow = (const v4f*)(s_up + (size_t)S.snd[lm] * MUL);
#pragma unroll
            for (int q = 0; q < 16; ++q) {
                v4f f = srow[q];
                row[q * 4 + 0] = (__bf16)f.x;
                row[q * 4 + 1] = (__bf16)f.y;
                row[q * 4 + 2] = (__bf16)f.z;
                row[q * 4 + 3] = (__bf16)f.w;
            }
            const v4f* rrow = (const v4f*)(radial + (size_t)es * NRAD);
#pragma unroll
            for (int q = 0; q < 8; ++q) {
                v4f f = rrow[q];
                row[128 + q * 4 + 0] = (__bf16)f.x;
                row[128 + q * 4 + 1] = (__bf16)f.y;
                row[128 + q * 4 + 2] = (__bf16)f.z;
                row[128 + q * 4 + 3] = (__bf16)f.w;
            }
        } else {
            // cols 64..127 : receiver scalars ; col 160 : length ; 161..191 : 0
            const v4f* srow = (const v4f*)(s_up + (size_t)S.rcv[lm] * MUL);
#pragma unroll
            for (int q = 0; q < 16; ++q) {
                v4f f = srow[q];
                row[64 + q * 4 + 0] = (__bf16)f.x;
                row[64 + q * 4 + 1] = (__bf16)f.y;
                row[64 + q * 4 + 2] = (__bf16)f.z;
                row[64 + q * 4 + 3] = (__bf16)f.w;
            }
            row[160] = (__bf16)lengths[es];
#pragma unroll
            for (int c = 161; c < K1P; ++c) row[c] = (__bf16)0.0f;
        }
    }
    __builtin_amdgcn_wave_barrier();

    // ---- Phase 3: layer 1 (161->128) --------------------------------------
    v8f acc[NT_H];
#pragma unroll
    for (int n = 0; n < NT_H; ++n) acc[n] = (v8f){0.f,0.f,0.f,0.f,0.f,0.f,0.f,0.f};
    for (int k = 0; k < K1_STEPS; ++k) {
        v16bf a = load_a_lds(S.buf, K1P, k, lane);
#pragma unroll
        for (int n = 0; n < NT_H; ++n) {
            v16bf b = load_frag_g(W1s, k * NT_H + n, lane);
            acc[n] = wmma_bf16(a, b, acc[n]);
        }
    }
#pragma unroll
    for (int n = 0; n < NT_H; ++n) {
        float bias = B1[n * 16 + lm];
#pragma unroll
        for (int r = 0; r < 8; ++r) {
            float h = silu(acc[n][r] + bias);
            // D layout: lane holds N=lm, M=r+8*lh  ->  store row-major, stride 128
            S.buf[(r + 8 * lh) * MLPW + n * 16 + lm] = (__bf16)h;
        }
    }
    __builtin_amdgcn_wave_barrier();

    // ---- Phase 4: layer 2 (128->128) --------------------------------------
#pragma unroll
    for (int n = 0; n < NT_H; ++n) acc[n] = (v8f){0.f,0.f,0.f,0.f,0.f,0.f,0.f,0.f};
    for (int k = 0; k < K2_STEPS; ++k) {
        v16bf a = load_a_lds(S.buf, MLPW, k, lane);
#pragma unroll
        for (int n = 0; n < NT_H; ++n) {
            v16bf b = load_frag_g(W2s, k * NT_H + n, lane);
            acc[n] = wmma_bf16(a, b, acc[n]);
        }
    }
#pragma unroll
    for (int n = 0; n < NT_H; ++n) {
        float bias = B2[n * 16 + lm];
#pragma unroll
        for (int r = 0; r < 8; ++r) {
            float h = silu(acc[n][r] + bias);
            S.buf[(r + 8 * lh) * MLPW + n * 16 + lm] = (__bf16)h;
        }
    }
    __builtin_amdgcn_wave_barrier();

    // ---- Phase 5: layer 3 (128->192), mix stays in D fragments ------------
    v8f mix[NT_M];
#pragma unroll
    for (int n = 0; n < NT_M; ++n) mix[n] = (v8f){0.f,0.f,0.f,0.f,0.f,0.f,0.f,0.f};
    for (int k = 0; k < K2_STEPS; ++k) {
        v16bf a = load_a_lds(S.buf, MLPW, k, lane);
#pragma unroll
        for (int n = 0; n < NT_M; ++n) {
            v16bf b = load_frag_g(W3s, k * NT_M + n, lane);
            mix[n] = wmma_bf16(a, b, mix[n]);
        }
    }
#pragma unroll
    for (int n = 0; n < NT_M; ++n) {
        float bias = B3[n * 16 + lm];
#pragma unroll
        for (int r = 0; r < 8; ++r) mix[n][r] += bias;
    }

    // ---- Phase 6: messages + atomic scatter -------------------------------
    const float inv_sqrt3 = 0.5773502691896258f;
    const float inv_sqrt2 = 0.7071067811865476f;
    const float invAvg    = 1.0f / 16.0f;   // 1 / AVG_NUM_NEIGHBORS

#pragma unroll
    for (int r = 0; r < 8; ++r) {
        int m = r + 8 * lh;                  // edge row handled by this lane
        int e = tile * 16 + m;
        if (e >= nEdges) continue;
        int sn = S.snd[m], rc = S.rcv[m];
        float ex = S.ev[m][0], ey = S.ev[m][1], ez = S.ev[m][2];
#pragma unroll
        for (int t = 0; t < 4; ++t) {
            int ch = t * 16 + lm;            // channel 0..63 for this lane
            const float* svp = v_up + ((size_t)sn * MUL + ch) * 3;
            float sx = svp[0], sy = svp[1], sz = svp[2];

            // m0: out0 * mix0   (mix0 col = ch -> ntile t)
            float out0 = (sx * ex + sy * ey + sz * ez) * inv_sqrt3;
            atomicAdd(&agg0[(size_t)rc * MUL + ch], out0 * mix[t][r] * invAvg);

            // o1b: cross(sv, e)/sqrt2 * mix1[64+ch]  (ntile 8+t)
            float mb = mix[8 + t][r] * inv_sqrt2 * invAvg;
            float cx = sy * ez - sz * ey;
            float cy = sz * ex - sx * ez;
            float cz = sx * ey - sy * ex;
            float* pb = agg1 + ((size_t)rc * (2 * MUL) + MUL + ch) * 3;
            atomicAdd(pb + 0, cx * mb);
            atomicAdd(pb + 1, cy * mb);
            atomicAdd(pb + 2, cz * mb);

            // o1a: ss[sn][ch] * e * mix1[ch]  (ntile 4+t)
            float ssv = s_up[(size_t)sn * MUL + ch];
            float ma  = mix[4 + t][r] * invAvg;
            float* pa = agg1 + ((size_t)rc * (2 * MUL) + ch) * 3;
            atomicAdd(pa + 0, ssv * ex * ma);
            atomicAdd(pa + 1, ssv * ey * ma);
            atomicAdd(pa + 2, ssv * ez * ma);
        }
    }
}

// ---------------------------------------------------------------------------
// Node down-projection + output concat. Block = 4 nodes x 64 channels.
// out[n, 0..63]         = agg0 @ W_dn_s + b
// out[n, 64 + d*3 + i]  = sum_c agg1[n,c,i] * W_dn_v[c,d]
// ---------------------------------------------------------------------------
__global__ void __launch_bounds__(256)
down_kernel(const float* __restrict__ agg0, const float* __restrict__ agg1,
            const float* __restrict__ Ws, const float* __restrict__ bs,
            const float* __restrict__ Wv, float* __restrict__ out, int nNodes) {
    __shared__ float a0[4][MUL];
    __shared__ float a1[4][2 * MUL][3];
    int ln  = threadIdx.x & 63;
    int sub = threadIdx.x >> 6;
    int node = blockIdx.x * 4 + sub;
    bool ok = node < nNodes;
    if (ok) {
        a0[sub][ln] = agg0[(size_t)node * MUL + ln];
#pragma unroll
        for (int t = 0; t < 2; ++t) {
            int c = ln + t * MUL;
            a1[sub][c][0] = agg1[((size_t)node * 2 * MUL + c) * 3 + 0];
            a1[sub][c][1] = agg1[((size_t)node * 2 * MUL + c) * 3 + 1];
            a1[sub][c][2] = agg1[((size_t)node * 2 * MUL + c) * 3 + 2];
        }
    }
    __syncthreads();
    if (!ok) return;
    float s = bs[ln];
    for (int c = 0; c < MUL; ++c) s += a0[sub][c] * Ws[c * MUL + ln];
    float v0 = 0.f, v1 = 0.f, v2 = 0.f;
    for (int c = 0; c < 2 * MUL; ++c) {
        float w = Wv[c * MUL + ln];
        v0 += a1[sub][c][0] * w;
        v1 += a1[sub][c][1] * w;
        v2 += a1[sub][c][2] * w;
    }
    size_t ob = (size_t)node * (MUL + 3 * MUL);
    out[ob + ln] = s;
    out[ob + MUL + ln * 3 + 0] = v0;
    out[ob + MUL + ln * 3 + 1] = v1;
    out[ob + MUL + ln * 3 + 2] = v2;
}

// ---------------------------------------------------------------------------
extern "C" void kernel_launch(void* const* d_in, const int* in_sizes, int n_in,
                              void* d_out, int out_size, void* d_ws, size_t ws_size,
                              hipStream_t stream) {
    const float* node_scalars = (const float*)d_in[0];
    const float* node_vectors = (const float*)d_in[1];
    const float* edge_vectors = (const float*)d_in[2];
    const float* radial       = (const float*)d_in[3];
    const float* lengths      = (const float*)d_in[4];
    const int*   senders      = (const int*)d_in[5];
    const int*   receivers    = (const int*)d_in[6];
    const float* W_up_s = (const float*)d_in[7];
    const float* b_up_s = (const float*)d_in[8];
    const float* W_up_v = (const float*)d_in[9];
    const float* W1     = (const float*)d_in[10];
    const float* b1     = (const float*)d_in[11];
    const float* W2     = (const float*)d_in[12];
    const float* b2     = (const float*)d_in[13];
    const float* W3     = (const float*)d_in[14];
    const float* b3     = (const float*)d_in[15];
    const float* W_dn_s = (const float*)d_in[16];
    const float* b_dn_s = (const float*)d_in[17];
    const float* W_dn_v = (const float*)d_in[18];

    const int nNodes = in_sizes[0] / MUL;
    const int nEdges = in_sizes[5];

    char* base = (char*)d_ws;
    size_t off = 0;
    auto alloc = [&](size_t bytes) -> void* {
        void* p = base + off;
        off = (off + bytes + 255) & ~(size_t)255;
        return p;
    };
    float*  s_up = (float*)alloc((size_t)nNodes * MUL * sizeof(float));
    float*  v_up = (float*)alloc((size_t)nNodes * MUL * 3 * sizeof(float));
    float*  agg0 = (float*)alloc((size_t)nNodes * MUL * sizeof(float));
    float*  agg1 = (float*)alloc((size_t)nNodes * 2 * MUL * 3 * sizeof(float));
    __bf16* W1s  = (__bf16*)alloc((size_t)FRAGS_L1 * 512 * sizeof(__bf16));
    __bf16* W2s  = (__bf16*)alloc((size_t)FRAGS_L2 * 512 * sizeof(__bf16));
    __bf16* W3s  = (__bf16*)alloc((size_t)FRAGS_L3 * 512 * sizeof(__bf16));

    // 1) weight swizzle (bf16 fragment-major)
    const int prepThreads = (FRAGS_L1 + FRAGS_L2 + FRAGS_L3) * 512;
    prep_weights<<<(prepThreads + 255) / 256, 256, 0, stream>>>(W1, W2, W3, W1s, W2s, W3s);

    // 2) zero accumulators (must happen every call: graph-replayed)
    size_t nz0 = (size_t)nNodes * MUL;
    size_t nz1 = (size_t)nNodes * 2 * MUL * 3;
    zero_kernel<<<(unsigned)((nz0 + 255) / 256), 256, 0, stream>>>(agg0, nz0);
    zero_kernel<<<(unsigned)((nz1 + 255) / 256), 256, 0, stream>>>(agg1, nz1);

    // 3) node up-projection
    up_kernel<<<(nNodes + 3) / 4, 256, 0, stream>>>(
        node_scalars, node_vectors, W_up_s, b_up_s, W_up_v, s_up, v_up, nNodes);

    // 4) edge MLP (WMMA) + messages + atomic aggregation
    const int nTiles = (nEdges + 15) / 16;
    edge_kernel<<<(nTiles + 7) / 8, 256, 0, stream>>>(
        senders, receivers, edge_vectors, radial, lengths, s_up, v_up,
        W1s, W2s, W3s, b1, b2, b3, agg0, agg1, nEdges);

    // 5) node down-projection + output
    down_kernel<<<(nNodes + 3) / 4, 256, 0, stream>>>(
        agg0, agg1, W_dn_s, b_dn_s, W_dn_v, (float*)d_out, nNodes);
}
